// CrossAttention_24275155157376
// MI455X (gfx1250) — matrix operand order repaired and checked by reading, not compiled, not run on hardware
//
#include <hip/hip_runtime.h>

typedef __bf16 bf16;
typedef __attribute__((ext_vector_type(16))) __bf16 bf16x16;
typedef __attribute__((ext_vector_type(8)))  __bf16 bf16x8;
typedef __attribute__((ext_vector_type(2)))  __bf16 bf16x2;
typedef __attribute__((ext_vector_type(8)))  float  f32x8;

#define B_  16
#define T_  1024
#define C_  584
#define H_  8
#define D_  73
#define CP  608   // C padded to multiple of 32 (19 K-chunks)
#define DP  96    // D padded to multiple of 32 (3 K-chunks)
#define WPROWS 672  // 7 blocks x 96 output cols for projection (rows >= 584 are zero)

static __device__ __forceinline__ f32x8 zero8() {
    f32x8 z;
#pragma unroll
    for (int i = 0; i < 8; ++i) z[i] = 0.0f;
    return z;
}

// fragment loader: two contiguous 16-byte halves
static __device__ __forceinline__ bf16x16 ld_frag(const bf16* p0, const bf16* p1) {
    bf16x8 lo = *(const bf16x8*)p0;
    bf16x8 hi = *(const bf16x8*)p1;
    bf16x16 r;
#pragma unroll
    for (int i = 0; i < 8; ++i) { r[i] = lo[i]; r[i + 8] = hi[i]; }
    return r;
}
static __device__ __forceinline__ bf16x16 ld_frag32(const bf16* p) { return ld_frag(p, p + 8); }

static __device__ __forceinline__ f32x8 wmma_bf16(bf16x16 a, bf16x16 b, f32x8 c) {
    return __builtin_amdgcn_wmma_f32_16x16x32_bf16(false, a, false, b, (short)0, c, false, false);
}

// ---------------------------------------------------------------------------
// Prep kernels: one-time f32 -> zero-padded bf16 conversions (div/mod-free).
// ---------------------------------------------------------------------------
// [B*T][584] f32 -> [B*T][608] bf16.  grid.x = row, 304 pairs per row.
__global__ __launch_bounds__(320)
void cvt_x_kernel(const float* __restrict__ src, bf16* __restrict__ dst) {
    const int row = blockIdx.x;
    const int c0  = threadIdx.x * 2;
    if (c0 >= CP) return;
    float f0 = (c0     < C_) ? src[row * C_ + c0]     : 0.0f;
    float f1 = (c0 + 1 < C_) ? src[row * C_ + c0 + 1] : 0.0f;
    bf16x2 v = { (bf16)f0, (bf16)f1 };
    *(bf16x2*)&dst[row * CP + c0] = v;
}

// Wq/Wk/Wv [H][584][73] f32 -> WT [3][H][96][608] bf16, transposed + padded.
// grid = (d=96, w*H+h=24), block = 608 (19 waves), tid = c.
__global__ __launch_bounds__(608)
void cvt_w_kernel(const float* __restrict__ Wq, const float* __restrict__ Wk,
                  const float* __restrict__ Wv, bf16* __restrict__ WT) {
    const int c = threadIdx.x;
    const int d = blockIdx.x;
    const int wh = blockIdx.y;             // w*H + h
    const int h = wh & (H_ - 1), w = wh >> 3;
    const float* W = (w == 0) ? Wq : (w == 1) ? Wk : Wv;
    float v = (d < D_ && c < C_) ? W[(h * C_ + c) * D_ + d] : 0.0f;
    WT[((size_t)wh * DP + d) * CP + c] = (bf16)v;
}

// Wp [584][584] f32 -> WpT [672][608] bf16, transposed + padded.
// grid.x = o (output col), block = 608, tid = c (input col / K dim).
__global__ __launch_bounds__(608)
void cvt_wp_kernel(const float* __restrict__ Wp, bf16* __restrict__ WpT) {
    const int c = threadIdx.x;
    const int o = blockIdx.x;
    float v = (o < C_ && c < C_) ? Wp[c * C_ + o] : 0.0f;
    WpT[(size_t)o * CP + c] = (bf16)v;
}

// ---------------------------------------------------------------------------
// Kernel 1: Q/K/V projection GEMM — LDS-free, barrier-free, branch-free loop.
// z=0: q = xb@WqT^T (scaled by C^-0.5*log2e), z=1: k = mb@WkT^T,
// z=2: v = xb@WvT^T stored transposed as vT[(bh*DP+d)*T + t].
// ---------------------------------------------------------------------------
__global__ __launch_bounds__(256)
void qkv_kernel(const bf16* __restrict__ xb, const bf16* __restrict__ mb,
                const bf16* __restrict__ WT,
                bf16* __restrict__ qb, bf16* __restrict__ kb, bf16* __restrict__ vT) {
    const int tid  = threadIdx.x;
    const int wave = tid >> 5, lane = tid & 31;
    const int g = lane >> 4, ln = lane & 15;
    const int rows0 = blockIdx.x * 128;
    const int b = rows0 / T_, t0 = rows0 % T_;
    const int h = blockIdx.y, which = blockIdx.z;

    const bf16* src  = (which == 1) ? mb : xb;
    const bf16* wt   = WT + (size_t)((which * H_ + h) * DP) * CP;
    const bf16* arow = src + (size_t)(rows0 + wave * 16 + ln) * CP;

    f32x8 acc[6];
#pragma unroll
    for (int nt = 0; nt < 6; ++nt) acc[nt] = zero8();

    for (int kk = 0; kk < CP; kk += 32) {
        bf16x16 a = ld_frag32(arow + kk + g * 16);
#pragma unroll
        for (int nt = 0; nt < 6; ++nt) {
            bf16x16 bf = ld_frag32(wt + (size_t)(nt * 16 + ln) * CP + kk + g * 16);
            acc[nt] = wmma_bf16(a, bf, acc[nt]);
        }
    }

    // fold softmax scale (in log2 domain) into q
    const float qs = (which == 0) ? (1.44269504089f * rsqrtf((float)C_)) : 1.0f;
    const int bh = b * H_ + h;
#pragma unroll
    for (int nt = 0; nt < 6; ++nt) {
#pragma unroll
        for (int r = 0; r < 8; ++r) {
            int t = t0 + wave * 16 + r + g * 8;
            int col = nt * 16 + ln;
            bf16 val = (bf16)(acc[nt][r] * qs);
            if (which == 2)      vT[(size_t)(bh * DP + col) * T_ + t] = val;
            else if (which == 1) kb[(size_t)(bh * T_ + t) * DP + col] = val;
            else                 qb[(size_t)(bh * T_ + t) * DP + col] = val;
        }
    }
}

// ---------------------------------------------------------------------------
// Kernel 2: flash attention.  16 query rows per wave, S in 32-wide chunks:
// 6 score WMMAs, online softmax (q pre-scaled to log2 domain), P staged via
// per-wave LDS tile for the layout swap, 6 P*V WMMAs.  No block barriers.
// ---------------------------------------------------------------------------
__global__ __launch_bounds__(256)
void attn_kernel(const bf16* __restrict__ qb, const bf16* __restrict__ kb,
                 const bf16* __restrict__ vT, bf16* __restrict__ attn) {
    __shared__ bf16 Ps[8][16][32];   // per-wave P tile

    const int tid  = threadIdx.x;
    const int wave = tid >> 5, lane = tid & 31;
    const int g = lane >> 4, ln = lane & 15;
    const int b = blockIdx.z, h = blockIdx.y;
    const int t0 = blockIdx.x * 128 + wave * 16;
    const int bh = b * H_ + h;

    bf16x16 qa[3];
#pragma unroll
    for (int kc = 0; kc < 3; ++kc) {
        const bf16* p = qb + (size_t)(bh * T_ + t0 + ln) * DP + kc * 32;
        qa[kc] = ld_frag32(p + g * 16);
    }

    f32x8 mreg, lreg, oacc[6];
#pragma unroll
    for (int r = 0; r < 8; ++r) { mreg[r] = -3.0e38f; lreg[r] = 0.0f; }
#pragma unroll
    for (int nt = 0; nt < 6; ++nt) oacc[nt] = zero8();

    const bf16* kbh = kb + (size_t)bh * T_ * DP;
    const bf16* vbh = vT + (size_t)bh * DP * T_;

    for (int s0 = 0; s0 < T_; s0 += 32) {
        f32x8 sa = zero8(), sb = zero8();
#pragma unroll
        for (int kc = 0; kc < 3; ++kc) {
            bf16x16 b0 = ld_frag32(&kbh[(s0 + ln) * DP + kc * 32 + g * 16]);
            bf16x16 b1 = ld_frag32(&kbh[(s0 + 16 + ln) * DP + kc * 32 + g * 16]);
            sa = wmma_bf16(qa[kc], b0, sa);
            sb = wmma_bf16(qa[kc], b1, sb);
        }
#pragma unroll
        for (int r = 0; r < 8; ++r) {
            float x0 = sa[r], x1 = sb[r];        // already in log2 domain
            float cm = fmaxf(x0, x1);
            cm = fmaxf(cm, __shfl_xor(cm, 1));
            cm = fmaxf(cm, __shfl_xor(cm, 2));
            cm = fmaxf(cm, __shfl_xor(cm, 4));
            cm = fmaxf(cm, __shfl_xor(cm, 8));
            float mold = mreg[r];
            float mnew = fmaxf(mold, cm);
            float corr = exp2f(mold - mnew);
            float p0 = exp2f(x0 - mnew), p1 = exp2f(x1 - mnew);
            float rs = p0 + p1;
            rs += __shfl_xor(rs, 1);
            rs += __shfl_xor(rs, 2);
            rs += __shfl_xor(rs, 4);
            rs += __shfl_xor(rs, 8);
            lreg[r] = lreg[r] * corr + rs;
            mreg[r] = mnew;
#pragma unroll
            for (int nt = 0; nt < 6; ++nt) oacc[nt][r] *= corr;
            Ps[wave][r + g * 8][ln]      = (bf16)p0;
            Ps[wave][r + g * 8][16 + ln] = (bf16)p1;
        }
        asm volatile("s_wait_dscnt 0" ::: "memory");   // in-wave DS RAW
        bf16x16 pa = ld_frag32(&Ps[wave][ln][g * 16]);
#pragma unroll
        for (int nt = 0; nt < 6; ++nt) {
            bf16x16 vb = ld_frag32(&vbh[(nt * 16 + ln) * T_ + s0 + g * 16]);
            oacc[nt] = wmma_bf16(pa, vb, oacc[nt]);
        }
    }

#pragma unroll
    for (int nt = 0; nt < 6; ++nt) {
#pragma unroll
        for (int r = 0; r < 8; ++r) {
            int d = nt * 16 + ln;
            if (d < D_) {
                int t = t0 + r + g * 8;
                attn[(size_t)(b * T_ + t) * CP + h * D_ + d] = (bf16)(oacc[nt][r] / lreg[r]);
            }
        }
    }
}

// ---------------------------------------------------------------------------
// Kernel 3: output projection  out = attn @ Wp + bp  — LDS-free, barrier-free.
// ---------------------------------------------------------------------------
__global__ __launch_bounds__(256)
void proj_kernel(const bf16* __restrict__ attn, const bf16* __restrict__ WpT,
                 const float* __restrict__ bp, float* __restrict__ out) {
    const int tid  = threadIdx.x;
    const int wave = tid >> 5, lane = tid & 31;
    const int g = lane >> 4, ln = lane & 15;
    const int rows0 = blockIdx.x * 128;
    const int col0  = blockIdx.y * 96;

    const bf16* arow = attn + (size_t)(rows0 + wave * 16 + ln) * CP;

    f32x8 acc[6];
#pragma unroll
    for (int nt = 0; nt < 6; ++nt) acc[nt] = zero8();

    for (int kk = 0; kk < CP; kk += 32) {
        bf16x16 a = ld_frag32(arow + kk + g * 16);
#pragma unroll
        for (int nt = 0; nt < 6; ++nt) {
            bf16x16 bf = ld_frag32(WpT + (size_t)(col0 + nt * 16 + ln) * CP + kk + g * 16);
            acc[nt] = wmma_bf16(a, bf, acc[nt]);
        }
    }

#pragma unroll
    for (int nt = 0; nt < 6; ++nt) {
#pragma unroll
        for (int r = 0; r < 8; ++r) {
            int col = col0 + nt * 16 + ln;
            if (col < C_) {
                int row = rows0 + wave * 16 + r + g * 8;
                out[(size_t)row * C_ + col] = acc[nt][r] + bp[col];
            }
        }
    }
}

// ---------------------------------------------------------------------------
extern "C" void kernel_launch(void* const* d_in, const int* in_sizes, int n_in,
                              void* d_out, int out_size, void* d_ws, size_t ws_size,
                              hipStream_t stream) {
    const float* x   = (const float*)d_in[0];
    const float* mem = (const float*)d_in[1];
    const float* Wq  = (const float*)d_in[2];
    const float* Wk  = (const float*)d_in[3];
    const float* Wv  = (const float*)d_in[4];
    const float* Wp  = (const float*)d_in[5];
    const float* bp  = (const float*)d_in[6];
    float* out = (float*)d_out;

    const size_t XB   = (size_t)B_ * T_ * CP * sizeof(bf16);       // 19,922,944
    const size_t QKV  = (size_t)B_ * H_ * T_ * DP * sizeof(bf16);  // 25,165,824
    const size_t WTB  = (size_t)3 * H_ * DP * CP * sizeof(bf16);   //  2,801,664
    char* ws = (char*)d_ws;
    size_t off = 0;
    bf16* xb   = (bf16*)(ws + off); off += XB;
    bf16* mb   = (bf16*)(ws + off); off += XB;
    bf16* qb   = (bf16*)(ws + off); off += QKV;
    bf16* kb   = (bf16*)(ws + off); off += QKV;
    bf16* vT   = (bf16*)(ws + off); off += QKV;
    bf16* attn = (bf16*)(ws + off); off += XB;
    bf16* WT   = (bf16*)(ws + off); off += WTB;
    bf16* WpT  = (bf16*)(ws + off);

    cvt_x_kernel <<<B_ * T_, 320, 0, stream>>>(x,   xb);
    cvt_x_kernel <<<B_ * T_, 320, 0, stream>>>(mem, mb);
    cvt_w_kernel <<<dim3(DP, 3 * H_), 608, 0, stream>>>(Wq, Wk, Wv, WT);
    cvt_wp_kernel<<<WPROWS, 608, 0, stream>>>(Wp, WpT);

    qkv_kernel<<<dim3(B_ * T_ / 128, H_, 3), 256, 0, stream>>>(xb, mb, WT, qb, kb, vT);
    hipMemsetAsync(attn, 0, XB, stream);   // zero pad columns 584..607
    attn_kernel<<<dim3(T_ / 128, H_, B_), 256, 0, stream>>>(qb, kb, vT, attn);
    proj_kernel<<<dim3(B_ * T_ / 128, 7), 256, 0, stream>>>(attn, WpT, bp, out);
}
